// AttractorState_26972394619235
// MI455X (gfx1250) — matrix operand order repaired and checked by reading, not compiled, not run on hardware
//
#include <hip/hip_runtime.h>
#include <hip/hip_bf16.h>

// ---------------------------------------------------------------------------
// AttractorState: C[b,s,d] = sum_t alpha^(S-1-t) * (hidden[b,t,:]@W[s,:] + bias[s]) * pos[b,t,d]
// Fused two-GEMM pipeline with bf16 WMMA (v_wmma_f32_16x16x32_bf16), wave32.
// ---------------------------------------------------------------------------

typedef __bf16 bf16_t;
typedef bf16_t v16bf __attribute__((ext_vector_type(16)));
typedef bf16_t v2bf  __attribute__((ext_vector_type(2)));
typedef float  v8f   __attribute__((ext_vector_type(8)));

#define S_LEN   8192
#define DM      512      // d_model
#define DS      512      // d_state
#define TSPLIT  8        // t-range splits (global f32 atomic reduce)
#define TCHUNK  32       // K-step of stage-2 GEMM per iteration
#define STILE   128      // d_state rows per workgroup
#define THREADS 512      // 16 waves (wave32)

#define HPT_STRIDE 40    // shorts; 80B rows -> 16B aligned b128 LDS loads, conflict-free
#define POS_STRIDE 40

union BF16Frag { v16bf v; unsigned int u[8]; uint4 q[2]; };

// native packed f32->bf16 conversion (v_cvt_pk_bf16_f32)
__device__ __forceinline__ unsigned int pkbf(float x, float y) {
    v2bf p;
    p.x = (bf16_t)x;
    p.y = (bf16_t)y;
    return __builtin_bit_cast(unsigned int, p);
}

__device__ __forceinline__ unsigned short bf1(float x) {
    return __builtin_bit_cast(unsigned short, (bf16_t)x);
}

__global__ void AttractorState_zero_kernel(float* __restrict__ p, int n) {
    int i = blockIdx.x * blockDim.x + threadIdx.x;
    if (i < n) p[i] = 0.0f;
}

__global__ __launch_bounds__(THREADS)
void AttractorState_26972394619235_kernel(const float* __restrict__ hs,   // (B,S,DM)
                                          const float* __restrict__ pe,   // (B,S,DM)
                                          const float* __restrict__ W,    // (DS,DM)
                                          const float* __restrict__ bias, // (DS,)
                                          float* __restrict__ out)        // (B,DS,DM)
{
    __shared__ alignas(16) unsigned short hpT [STILE * HPT_STRIDE]; // [s][t] bf16
    __shared__ alignas(16) unsigned short posT[DM    * POS_STRIDE]; // [d][t] bf16

    const int tid  = threadIdx.x;
    const int lane = tid & 31;
    const int wave = tid >> 5;
    const int h    = lane >> 4;   // lane half (K-half select)
    const int m    = lane & 15;   // M/N index within a 16-wide tile

    const int stile  = blockIdx.x & 3;
    const int batch  = (blockIdx.x >> 2) & 3;
    const int tsplit = blockIdx.x >> 4;

    const int sbase   = stile * STILE;
    const int tbase   = tsplit * (S_LEN / TSPLIT);
    const int nchunks = (S_LEN / TSPLIT) / TCHUNK;

    // Stage-1 tile ownership: 2 t-subtiles x 8 s-subtiles = 16 waves
    const int tt = wave & 1;
    const int ss = wave >> 1;
    // Stage-2 ownership: 4 s-groups (32 rows) x 4 d-groups (128 cols)
    const int sgroup = wave >> 2;
    const int dgroup = wave & 3;

    v8f acc[2][8];
#pragma unroll
    for (int si = 0; si < 2; ++si)
#pragma unroll
        for (int dj = 0; dj < 8; ++dj) acc[si][dj] = (v8f){};

    const float C0     = 3.8349519697141028e-4f; // pi / S
    const float bias_v = bias[sbase + ss * 16 + m];

    // loop-invariant part of the decay: exp(C0 * t_local) per accumulator row
    float wtl[8];
#pragma unroll
    for (int r = 0; r < 8; ++r)
        wtl[r] = __expf(C0 * (float)(tt * 16 + r + 8 * h));

    const float* Abase = hs + ((size_t)batch * S_LEN) * DM;
    const float* Brow  = W  + (size_t)(sbase + ss * 16 + m) * DM;

    for (int chunk = 0; chunk < nchunks; ++chunk) {
        const int t0 = tbase + chunk * TCHUNK;
        __syncthreads(); // previous chunk's LDS readers done

        // ---- cooperative transpose-stage of pos chunk: pe[b][t0+t][d] -> posT[d][t]
#pragma unroll
        for (int i = 0; i < (TCHUNK * DM) / THREADS; ++i) {
            int idx = tid + i * THREADS;
            int t   = idx >> 9;        // /512 (DM)
            int d   = idx & (DM - 1);
            float v = pe[((size_t)batch * S_LEN + (t0 + t)) * DM + d];
            posT[d * POS_STRIDE + t] = bf1(v);
        }

        // ---- stage 1: hp tile (16t x 16s), K = DM = 512, 16 WMMAs
        v8f hpacc = (v8f){};
        const float* arow = Abase + (size_t)(t0 + tt * 16 + m) * DM;
        for (int k = 0; k < DM; k += 32) {
            // A (hidden, 16x32): K-pairs are two 16B-contiguous runs per lane
            float4 fa0 = *(const float4*)(arow + k + 8 * h);
            float4 fa1 = *(const float4*)(arow + k + 8 * h + 4);
            float4 fa2 = *(const float4*)(arow + k + 16 + 8 * h);
            float4 fa3 = *(const float4*)(arow + k + 16 + 8 * h + 4);
            // B (W, 32x16): K-pairs are two 16B-contiguous runs per lane
            float4 fb0 = *(const float4*)(Brow + k + 8 * h);
            float4 fb1 = *(const float4*)(Brow + k + 8 * h + 4);
            float4 fb2 = *(const float4*)(Brow + k + 16 + 8 * h);
            float4 fb3 = *(const float4*)(Brow + k + 16 + 8 * h + 4);

            BF16Frag a, b;
            a.u[0] = pkbf(fa0.x, fa0.y); a.u[1] = pkbf(fa0.z, fa0.w);
            a.u[2] = pkbf(fa1.x, fa1.y); a.u[3] = pkbf(fa1.z, fa1.w);
            a.u[4] = pkbf(fa2.x, fa2.y); a.u[5] = pkbf(fa2.z, fa2.w);
            a.u[6] = pkbf(fa3.x, fa3.y); a.u[7] = pkbf(fa3.z, fa3.w);
            b.u[0] = pkbf(fb0.x, fb0.y); b.u[1] = pkbf(fb0.z, fb0.w);
            b.u[2] = pkbf(fb1.x, fb1.y); b.u[3] = pkbf(fb1.z, fb1.w);
            b.u[4] = pkbf(fb2.x, fb2.y); b.u[5] = pkbf(fb2.z, fb2.w);
            b.u[6] = pkbf(fb3.x, fb3.y); b.u[7] = pkbf(fb3.z, fb3.w);

            hpacc = __builtin_amdgcn_wmma_f32_16x16x32_bf16(
                false, a.v, false, b.v, (short)0, hpacc, false, false);
        }

        // epilogue: + bias, * alpha^(S-1-t) (one uniform exp per chunk), store transposed
        const float wbase = __expf(C0 * ((float)t0 - (float)(S_LEN - 1)));
#pragma unroll
        for (int r = 0; r < 8; ++r) {
            int tl = tt * 16 + r + 8 * h;  // t within chunk (D-matrix M)
            int sl = ss * 16 + m;          // s within STILE (D-matrix N)
            float val = (hpacc[r] + bias_v) * (wbase * wtl[r]);
            hpT[sl * HPT_STRIDE + tl] = bf1(val);
        }
        __syncthreads(); // hpT + posT visible to all waves

        // ---- stage 2: C[32s x 128d] += hpT @ posT^T, K = 32, 16 WMMAs
        BF16Frag afr[2];
#pragma unroll
        for (int si = 0; si < 2; ++si) {
            const unsigned short* p = &hpT[(sgroup * 32 + si * 16 + m) * HPT_STRIDE];
            afr[si].q[0] = *(const uint4*)(p + 8 * h);       // A pairs, low run
            afr[si].q[1] = *(const uint4*)(p + 16 + 8 * h);  // A pairs, high run
        }
#pragma unroll
        for (int dj = 0; dj < 8; ++dj) {
            const unsigned short* p = &posT[(dgroup * 128 + dj * 16 + m) * POS_STRIDE];
            BF16Frag bfr;
            bfr.q[0] = *(const uint4*)(p + 8 * h);           // B pairs, low run
            bfr.q[1] = *(const uint4*)(p + 16 + 8 * h);
#pragma unroll
            for (int si = 0; si < 2; ++si)
                acc[si][dj] = __builtin_amdgcn_wmma_f32_16x16x32_bf16(
                    false, afr[si].v, false, bfr.v, (short)0, acc[si][dj], false, false);
        }
    }

    // ---- reduce partial C tile into global output (TSPLIT-way f32 atomic add)
#pragma unroll
    for (int si = 0; si < 2; ++si) {
#pragma unroll
        for (int r = 0; r < 8; ++r) {
            int s = sbase + sgroup * 32 + si * 16 + r + 8 * h;
#pragma unroll
            for (int dj = 0; dj < 8; ++dj) {
                int d = dgroup * 128 + dj * 16 + m;
                atomicAdd(&out[((size_t)batch * DS + s) * DM + d], acc[si][dj][r]);
            }
        }
    }
}

extern "C" void kernel_launch(void* const* d_in, const int* in_sizes, int n_in,
                              void* d_out, int out_size, void* d_ws, size_t ws_size,
                              hipStream_t stream) {
    const float* hs   = (const float*)d_in[0];
    const float* pe   = (const float*)d_in[1];
    const float* W    = (const float*)d_in[2];
    const float* bias = (const float*)d_in[3];
    float* out = (float*)d_out;

    const int n = 4 * DS * DM;
    AttractorState_zero_kernel<<<(n + 255) / 256, 256, 0, stream>>>(out, n);

    dim3 grid(4 /*s-tiles*/ * 4 /*batches*/ * TSPLIT);
    AttractorState_26972394619235_kernel<<<grid, THREADS, 0, stream>>>(hs, pe, W, bias, out);
}
